// Codebook_58428735095071
// MI455X (gfx1250) — matrix operand order repaired and checked by reading, not compiled, not run on hardware
//
#include <hip/hip_runtime.h>

typedef __attribute__((ext_vector_type(4)))  float     v4f;
typedef __attribute__((ext_vector_type(8)))  float     v8f;
typedef __attribute__((ext_vector_type(8)))  _Float16  v8h;
typedef __attribute__((ext_vector_type(16))) _Float16  v16h;

#define D_DIM 256
#define K_CODES 1024
#define ROWS_PER_BLK 64
#define THREADS 128
#define TILE_COLS 32                    // codes per pipeline stage
#define N_TILES (K_CODES / TILE_COLS)   // 32
#define BROW 264                        // padded halves per LDS B row (528B: 16B-aligned, stride%64dw=4 -> conflict-free)

// ---------------------------------------------------------------------------
// Prep: emb -> f16 copy (for WMMA filter) + fp32 ||e||^2 (for distances)
// ---------------------------------------------------------------------------
__global__ __launch_bounds__(256) void prep_kernel(const float* __restrict__ emb,
                                                   _Float16* __restrict__ emb_h,
                                                   float* __restrict__ enorm) {
  __shared__ float wsum[8];
  const int k = blockIdx.x;
  const int d = threadIdx.x;
  float f = emb[k * D_DIM + d];
  emb_h[k * D_DIM + d] = (_Float16)f;
  float s = f * f;
#pragma unroll
  for (int m = 16; m >= 1; m >>= 1) s += __shfl_xor(s, m, 32);
  if ((d & 31) == 0) wsum[d >> 5] = s;
  __syncthreads();
  if (d == 0) {
    float t = 0.0f;
#pragma unroll
    for (int i = 0; i < 8; ++i) t += wsum[i];
    enorm[k] = t;
  }
}

// ---------------------------------------------------------------------------
// Fused VQ kernel: f16 WMMA filter (top-2 per row) + fp32 re-rank + outputs.
// Block: 128 threads / 4 waves, 64 rows. Wave w owns rows [w*16, w*16+16).
// B tiles (32 codes, f16) staged in LDS via async-to-LDS, triple-buffered.
// ---------------------------------------------------------------------------
__global__ __launch_bounds__(THREADS) void vq_kernel(const float* __restrict__ x,
                                                     const float* __restrict__ emb,
                                                     const _Float16* __restrict__ emb_h,
                                                     const float* __restrict__ enorm,
                                                     float* __restrict__ xq,
                                                     float* __restrict__ p) {
  __shared__ _Float16 btile[3][TILE_COLS * BROW];   // ~50.7 KB
  __shared__ int cands[4][16][2];
  __shared__ int sidx[ROWS_PER_BLK];

  const int tid  = threadIdx.x;
  const int wave = tid >> 5;
  const int lane = tid & 31;
  const int nlo  = lane & 15;               // A: M row; B/C: N col
  const int hi   = lane >> 4;
  const long row_base = (long)blockIdx.x * ROWS_PER_BLK;

  // ---- A fragments: wave's 16x256 x tile as f16, in registers (64 VGPRs) ----
  // 16-bit A 16x32 order used here (not AMD's table order): lo lanes K=0..15 contiguous, hi lanes K=16..31 (matches B)
  const int akoff = hi * 16;
  v16h areg[8];
  {
    const float* xr = x + (row_base + wave * 16 + nlo) * D_DIM;
#pragma unroll
    for (int kb = 0; kb < 8; ++kb) {
      v4f f0 = *(const v4f*)(xr + kb * 32 + akoff);
      v4f f1 = *(const v4f*)(xr + kb * 32 + akoff + 4);
      v4f f2 = *(const v4f*)(xr + kb * 32 + akoff + 8);
      v4f f3 = *(const v4f*)(xr + kb * 32 + akoff + 12);
      v16h a;
#pragma unroll
      for (int j = 0; j < 4; ++j) {
        a[j]      = (_Float16)f0[j];
        a[4 + j]  = (_Float16)f1[j];
        a[8 + j]  = (_Float16)f2[j];
        a[12 + j] = (_Float16)f3[j];
      }
      areg[kb] = a;
    }
  }

  float bv1[8], bv2[8];
  int   bi1[8], bi2[8];
#pragma unroll
  for (int i = 0; i < 8; ++i) { bv1[i] = 3.0e38f; bv2[i] = 3.0e38f; bi1[i] = 0; bi2[i] = 0; }

  const unsigned lds0 = (unsigned)(size_t)(&btile[0][0]);

  // Async-copy tile t into buffer t%3: 32 rows x 512B; wave copies 8 rows (8 x b128/lane ops).
  auto issue_copy = [&](int t) {
    const int b = t % 3;
    const unsigned lbase = lds0 + (unsigned)(b * TILE_COLS * BROW * 2);
    const _Float16* src = emb_h + (size_t)t * TILE_COLS * D_DIM;
#pragma unroll
    for (int j = 0; j < 8; ++j) {
      const int r = wave * 8 + j;
      const unsigned la = lbase + (unsigned)(r * (BROW * 2) + lane * 16);
      const unsigned vo = (unsigned)(r * 512 + lane * 16);
      asm volatile("global_load_async_to_lds_b128 %0, %1, %2"
                   :: "v"(la), "v"(vo), "s"(src)
                   : "memory");
    }
  };

  // Branch-free top-2 update (pure v_cndmask, no exec juggling, no scratch).
  auto upd = [&](int i, float val, int col) {
    const bool  lt1 = val < bv1[i];
    const bool  lt2 = val < bv2[i];
    const float pv1 = bv1[i];
    const int   pi1 = bi1[i];
    bv1[i] = lt1 ? val : pv1;
    bi1[i] = lt1 ? col : pi1;
    bv2[i] = lt1 ? pv1 : (lt2 ? val : bv2[i]);
    bi2[i] = lt1 ? pi1 : (lt2 ? col : bi2[i]);
  };

  // Filter: scores for tile t from LDS, update per-slot top-2.
  // 16-bit B 32x16 order used here (not AMD's table order): lo lanes K=0..15 contiguous, hi lanes K=16..31.
  // Two interleaved accumulator chains + B fragments double-buffered 1 kb ahead
  // so dscnt waits are pre-satisfied when the wmma issues.
  const int bkoff = hi * 16;
  union HU { v16h v; v8h h[2]; };
  auto compute = [&](int t) {
    const _Float16* bb    = &btile[t % 3][0];
    const _Float16* brow0 = bb + nlo * BROW;
    const _Float16* brow1 = bb + (16 + nlo) * BROW;

    v8f c0 = {}, c1 = {};
    HU b0[2], b1[2];
    b0[0].h[0] = *(const v8h*)(brow0 + bkoff);
    b0[0].h[1] = *(const v8h*)(brow0 + bkoff + 8);
    b1[0].h[0] = *(const v8h*)(brow1 + bkoff);
    b1[0].h[1] = *(const v8h*)(brow1 + bkoff + 8);
#pragma unroll
    for (int kb = 0; kb < 8; ++kb) {
      const int cur = kb & 1, nxt = cur ^ 1;
      if (kb < 7) {
        b0[nxt].h[0] = *(const v8h*)(brow0 + (kb + 1) * 32 + bkoff);
        b0[nxt].h[1] = *(const v8h*)(brow0 + (kb + 1) * 32 + bkoff + 8);
        b1[nxt].h[0] = *(const v8h*)(brow1 + (kb + 1) * 32 + bkoff);
        b1[nxt].h[1] = *(const v8h*)(brow1 + (kb + 1) * 32 + bkoff + 8);
      }
      c0 = __builtin_amdgcn_wmma_f32_16x16x32_f16(false, areg[kb], false, b0[cur].v,
                                                  (short)0, c0, false, false);
      c1 = __builtin_amdgcn_wmma_f32_16x16x32_f16(false, areg[kb], false, b1[cur].v,
                                                  (short)0, c1, false, false);
    }
    const int col0 = t * TILE_COLS + nlo;
    const int col1 = col0 + 16;
    const float en0 = enorm[col0];
    const float en1 = enorm[col1];
#pragma unroll
    for (int i = 0; i < 8; ++i) {
      upd(i, fmaf(-2.0f, c0[i], en0), col0);
      upd(i, fmaf(-2.0f, c1[i], en1), col1);
    }
  };

  // ---- pipeline: prefetch 1 tile ahead, triple buffer, 1 barrier/tile ----
  issue_copy(0);
#pragma unroll 1
  for (int t = 0; t < N_TILES - 1; ++t) {
    issue_copy(t + 1);
    asm volatile("s_wait_asynccnt 8" ::: "memory");   // copy(t) done (copy(t+1)'s 8 may remain)
    __syncthreads();
    compute(t);
  }
  asm volatile("s_wait_asynccnt 0" ::: "memory");
  __syncthreads();
  compute(N_TILES - 1);

  // ---- butterfly top-2 merge across the 16-lane N groups (branch-free) ----
#pragma unroll
  for (int mask = 1; mask < 16; mask <<= 1) {
#pragma unroll
    for (int i = 0; i < 8; ++i) {
      const float o1 = __shfl_xor(bv1[i], mask, 32);
      const int   q1 = __shfl_xor(bi1[i], mask, 32);
      const float o2 = __shfl_xor(bv2[i], mask, 32);
      const int   q2 = __shfl_xor(bi2[i], mask, 32);
      const bool  f  = o1 < bv1[i];
      const float w1 = f ? o1 : bv1[i];     // overall best
      const int   wi = f ? q1 : bi1[i];
      const float ls = f ? bv1[i] : o1;     // loser of the firsts
      const int   li = f ? bi1[i] : q1;
      const float ws = f ? o2 : bv2[i];     // winner's own second
      const int   wsi = f ? q2 : bi2[i];
      const bool  g  = ws < ls;
      bv1[i] = w1;               bi1[i] = wi;
      bv2[i] = g ? ws : ls;      bi2[i] = g ? wsi : li;
    }
  }
  // lane 0 holds rows 0..7 (C VGPR i -> M=i), lane 16 rows 8..15
  if (nlo == 0) {
    const int rb = hi * 8;
#pragma unroll
    for (int i = 0; i < 8; ++i) {
      cands[wave][rb + i][0] = bi1[i];
      cands[wave][rb + i][1] = bi2[i];
    }
  }

  // ---- fp32 re-rank: 32 lanes = 16 rows x 2 candidates ----
  {
    const int r  = lane >> 1;
    const int ci = cands[wave][r][lane & 1];
    const float* xr = x   + (row_base + wave * 16 + r) * D_DIM;
    const float* er = emb + (size_t)ci * D_DIM;
    float dot = 0.0f;
#pragma unroll 8
    for (int j = 0; j < 64; ++j) {
      v4f xa = *(const v4f*)(xr + 4 * j);
      v4f ea = *(const v4f*)(er + 4 * j);
      dot += xa[0] * ea[0] + xa[1] * ea[1] + xa[2] * ea[2] + xa[3] * ea[3];
    }
    const float dv = enorm[ci] - 2.0f * dot;
    const float ov = __shfl_xor(dv, 1, 32);
    const int   oi = __shfl_xor(ci, 1, 32);
    const int fin = (ov < dv || (ov == dv && oi < ci)) ? oi : ci;
    if ((lane & 1) == 0) sidx[wave * 16 + r] = fin;
  }

  // ---- outputs (per-wave rows; same-wave LDS dependence, no barrier) ----
  // xq = emb[idx]
  for (int j = lane; j < 16 * (D_DIM / 4); j += 32) {
    const int m = j >> 6, d4 = j & 63;
    const int idx = sidx[wave * 16 + m];
    v4f v = *(const v4f*)(emb + (size_t)idx * D_DIM + d4 * 4);
    *(v4f*)(xq + (row_base + wave * 16 + m) * D_DIM + d4 * 4) = v;
  }
  // p = one_hot(idx): zero b128 stream with 1.0 injected in owning chunk
  for (int j = lane; j < 16 * (K_CODES / 4); j += 32) {
    const int m = j >> 8, c4 = j & 255;
    const int fi = sidx[wave * 16 + m];
    v4f v = {0.0f, 0.0f, 0.0f, 0.0f};
    if ((fi >> 2) == c4) v[fi & 3] = 1.0f;
    *(v4f*)(p + (row_base + wave * 16 + m) * (long)K_CODES + c4 * 4) = v;
  }
}

// ---------------------------------------------------------------------------
extern "C" void kernel_launch(void* const* d_in, const int* in_sizes, int n_in,
                              void* d_out, int out_size, void* d_ws, size_t ws_size,
                              hipStream_t stream) {
  const float* x   = (const float*)d_in[0];   // [N, 256]
  const float* emb = (const float*)d_in[1];   // [1024, 256]

  const int N = in_sizes[0] / D_DIM;          // 65536
  const int K = in_sizes[1] / D_DIM;          // 1024

  float* xq = (float*)d_out;                  // [N, 256]
  float* p  = xq + (size_t)N * D_DIM;         // [N, 1024]

  _Float16* emb_h = (_Float16*)d_ws;                                   // 512 KB
  float*    enorm = (float*)((char*)d_ws + (size_t)K * D_DIM * 2);     // 4 KB

  prep_kernel<<<K, 256, 0, stream>>>(emb, emb_h, enorm);
  vq_kernel<<<N / ROWS_PER_BLK, THREADS, 0, stream>>>(x, emb, emb_h, enorm, xq, p);
}
